// MaskedLinear_61040075210832
// MI455X (gfx1250) — compile-verified, matrix operand order checked by reading, not run on hardware
//
#include <hip/hip_runtime.h>
#include <hip/hip_bf16.h>

typedef __bf16 bhalf_t;
typedef __attribute__((ext_vector_type(16))) __bf16 v16bf;
typedef __attribute__((ext_vector_type(8)))  __bf16 v8bf;
typedef __attribute__((ext_vector_type(8)))  float  v8f;

constexpr int Bn = 8, Ln = 2048, Dn = 1024, Hn = 16, dh = 64;

// ---------------------------------------------------------------------------
// Kernel 0a: transpose + convert x (B,L,D) f32 -> xT rows (b*D + c) of length L (bf16)
// so that WMMA B-fragments read contiguous K (=l) runs.
// ---------------------------------------------------------------------------
__global__ void k_xpose(const float* __restrict__ x, bhalf_t* __restrict__ xT) {
  __shared__ float tile[32][33];
  const int b  = blockIdx.z;
  const int l0 = blockIdx.x * 32, c0 = blockIdx.y * 32;
  const int tx = threadIdx.x, ty = threadIdx.y;
  tile[ty][tx] = x[((size_t)b * Ln + (l0 + ty)) * Dn + (c0 + tx)];
  __syncthreads();
  xT[((size_t)b * Dn + (c0 + ty)) * Ln + (l0 + tx)] = (bhalf_t)tile[tx][ty];
}

// Kernel 0b: convert W_proj (D,D) f32 -> bf16 (row-major; rows are the GEMM-2 N dim,
// cols the K dim, so B-fragments read contiguous K runs).
__global__ void k_wproj(const float* __restrict__ Wp, bhalf_t* __restrict__ Wpb) {
  const int i = blockIdx.x * 256 + threadIdx.x;
  Wpb[i] = (bhalf_t)Wp[i];
}

// ---------------------------------------------------------------------------
// Kernel 1: per-(head,row) softmax stats over the masked row.
// Masked (l>i) entries are value 0 (tril multiplies the weight), so they DO
// contribute exp(0) to the sum — faithful to the reference.
// ---------------------------------------------------------------------------
__global__ void __launch_bounds__(256) k_stats(const float* __restrict__ W,
                                               float* __restrict__ row_m,
                                               float* __restrict__ row_inv) {
  const int row = blockIdx.x;          // h*L + i
  const int i   = row & (Ln - 1);
  const float* wr = W + (size_t)row * Ln;
  const int tid = threadIdx.x;

  float v[8];
  float mx = -3.4e38f;
#pragma unroll
  for (int j = 0; j < 8; ++j) {
    const int l = tid + j * 256;
    const float w = wr[l];
    v[j] = (l <= i) ? w : 0.0f;
    mx = fmaxf(mx, v[j]);
  }
  __shared__ float red[256];
  red[tid] = mx;
  __syncthreads();
  for (int s = 128; s > 0; s >>= 1) {
    if (tid < s) red[tid] = fmaxf(red[tid], red[tid + s]);
    __syncthreads();
  }
  const float m = red[0];
  __syncthreads();

  float sum = 0.0f;
#pragma unroll
  for (int j = 0; j < 8; ++j) sum += __expf(v[j] - m);
  red[tid] = sum;
  __syncthreads();
  for (int s = 128; s > 0; s >>= 1) {
    if (tid < s) red[tid] += red[tid + s];
    __syncthreads();
  }
  if (tid == 0) { row_m[row] = m; row_inv[row] = 1.0f / red[0]; }
}

// ---------------------------------------------------------------------------
// Kernel 2: position mixing. For head h, row tile i0..i0+63:
//   mixed[b, i, h*64+dd] = sum_l softmax(W*mask)[h,i,l] * x[b,l,h*64+dd]
// A-fragment (normalized weights) is built on the fly from fp32 W using the
// precomputed row stats; one A fragment feeds 4 batches x 4 d-tiles of WMMA.
// Per-lane ISA layout (16-bit A/B, wave32): row/col = lane%16,
//   K(e) = (e<8 ? 0 : 16) + (lane>=16 ? 8 : 0) + (e&7).
// ---------------------------------------------------------------------------
__global__ void __launch_bounds__(256) k_mix(const float* __restrict__ W,
                                             const bhalf_t* __restrict__ xT,
                                             const float* __restrict__ row_m,
                                             const float* __restrict__ row_inv,
                                             bhalf_t* __restrict__ mix) {
  const int h    = blockIdx.y;
  const int i0   = blockIdx.x * 64;
  const int tid  = threadIdx.x;
  const int wave = tid >> 5;
  const int lane = tid & 31;
  const int rt   = wave & 3;          // row sub-tile (16 rows)
  const int bg   = (wave >> 2) * 4;   // batch group base
  const int lm   = lane & 15;
  const int hi   = lane >> 4;         // 0/1 half-wave
  const int c0   = hi * 8;

  const int   i    = i0 + rt * 16 + lm;            // A row owned by this lane
  const float mrow = row_m[h * Ln + i];
  const float irow = row_inv[h * Ln + i];
  const float* wr  = W + ((size_t)h * Ln + i) * Ln;

  // B source rows: xT row = b*D + h*64 + dt*16 + lm  (contiguous in K=l)
  const bhalf_t* bp[4][4];
#pragma unroll
  for (int bb = 0; bb < 4; ++bb)
#pragma unroll
    for (int dt = 0; dt < 4; ++dt)
      bp[bb][dt] = xT + ((size_t)(bg + bb) * Dn + h * dh + dt * 16 + lm) * Ln;

  v8f acc[4][4];
#pragma unroll
  for (int bb = 0; bb < 4; ++bb)
#pragma unroll
    for (int dt = 0; dt < 4; ++dt) acc[bb][dt] = v8f{0, 0, 0, 0, 0, 0, 0, 0};

  for (int kk = 0; kk < Ln; kk += 32) {
    // ---- A fragment: masked, exp-normalized W row (fp32 -> bf16)
    const float4 f0 = *(const float4*)(wr + kk + c0);
    const float4 f1 = *(const float4*)(wr + kk + c0 + 4);
    const float4 f2 = *(const float4*)(wr + kk + 16 + c0);
    const float4 f3 = *(const float4*)(wr + kk + 16 + c0 + 4);
    const float fa[16] = {f0.x, f0.y, f0.z, f0.w, f1.x, f1.y, f1.z, f1.w,
                          f2.x, f2.y, f2.z, f2.w, f3.x, f3.y, f3.z, f3.w};
    v16bf a;
#pragma unroll
    for (int e = 0; e < 16; ++e) {
      const int   k   = kk + ((e < 8) ? 0 : 16) + c0 + (e & 7);
      const float val = (k <= i) ? fa[e] : 0.0f;   // tril multiplies the weight
      a[e] = (bhalf_t)(__expf(val - mrow) * irow);
    }
#pragma unroll
    for (int bb = 0; bb < 4; ++bb) {
#pragma unroll
      for (int dt = 0; dt < 4; ++dt) {
        const v8bf blo = *(const v8bf*)(bp[bb][dt] + kk + c0);
        const v8bf bhi = *(const v8bf*)(bp[bb][dt] + kk + 16 + c0);
        v16bf bfrag;
#pragma unroll
        for (int e = 0; e < 8; ++e) { bfrag[e] = blo[e]; bfrag[8 + e] = bhi[e]; }
        acc[bb][dt] = __builtin_amdgcn_wmma_f32_16x16x32_bf16(
            false, a, false, bfrag, (short)0, acc[bb][dt], false, false);
      }
    }
  }

  // store mixed as bf16, row = b*L + i_out, col = h*64 + dt*16 + lm
#pragma unroll
  for (int bb = 0; bb < 4; ++bb) {
#pragma unroll
    for (int dt = 0; dt < 4; ++dt) {
#pragma unroll
      for (int r = 0; r < 8; ++r) {
        const int io = i0 + rt * 16 + r + hi * 8;   // C layout: M = r + 8*hi
        const size_t idx =
            ((size_t)(bg + bb) * Ln + io) * Dn + h * dh + dt * 16 + lm;
        mix[idx] = (bhalf_t)acc[bb][dt][r];
      }
    }
  }
}

// ---------------------------------------------------------------------------
// Kernel 3: projection GEMM  out(16384x1024) = mix @ W_proj^T + b_proj  (fp32 out)
// B[k][n] = W_proj[n][k] -> per-lane column fragment reads W_proj row n contiguously.
// ---------------------------------------------------------------------------
__global__ void __launch_bounds__(256) k_proj(const bhalf_t* __restrict__ mix,
                                              const bhalf_t* __restrict__ Wpb,
                                              const float* __restrict__ bias,
                                              float* __restrict__ out) {
  const int m0   = blockIdx.x * 64;
  const int n0   = blockIdx.y * 64;
  const int tid  = threadIdx.x;
  const int wave = tid >> 5, lane = tid & 31;
  const int rt   = wave & 3;
  const int ct0  = (wave >> 2) * 2;
  const int lm   = lane & 15, hi = lane >> 4, c0 = hi * 8;

  const bhalf_t* ap  = mix + (size_t)(m0 + rt * 16 + lm) * Dn;
  const bhalf_t* bp0 = Wpb + (size_t)(n0 + ct0 * 16 + lm) * Dn;
  const bhalf_t* bp1 = Wpb + (size_t)(n0 + (ct0 + 1) * 16 + lm) * Dn;

  v8f acc0 = v8f{0, 0, 0, 0, 0, 0, 0, 0};
  v8f acc1 = v8f{0, 0, 0, 0, 0, 0, 0, 0};

  for (int kk = 0; kk < Dn; kk += 32) {
    const v8bf alo = *(const v8bf*)(ap + kk + c0);
    const v8bf ahi = *(const v8bf*)(ap + kk + 16 + c0);
    const v8bf b0l = *(const v8bf*)(bp0 + kk + c0);
    const v8bf b0h = *(const v8bf*)(bp0 + kk + 16 + c0);
    const v8bf b1l = *(const v8bf*)(bp1 + kk + c0);
    const v8bf b1h = *(const v8bf*)(bp1 + kk + 16 + c0);
    v16bf a, b0, b1;
#pragma unroll
    for (int e = 0; e < 8; ++e) {
      a[e] = alo[e];  a[8 + e] = ahi[e];
      b0[e] = b0l[e]; b0[8 + e] = b0h[e];
      b1[e] = b1l[e]; b1[8 + e] = b1h[e];
    }
    acc0 = __builtin_amdgcn_wmma_f32_16x16x32_bf16(false, a, false, b0, (short)0,
                                                   acc0, false, false);
    acc1 = __builtin_amdgcn_wmma_f32_16x16x32_bf16(false, a, false, b1, (short)0,
                                                   acc1, false, false);
  }

  const int   col0 = n0 + ct0 * 16 + lm;
  const int   col1 = col0 + 16;
  const float bia0 = bias[col0], bia1 = bias[col1];
#pragma unroll
  for (int r = 0; r < 8; ++r) {
    const int row = m0 + rt * 16 + r + hi * 8;
    out[(size_t)row * Dn + col0] = acc0[r] + bia0;
    out[(size_t)row * Dn + col1] = acc1[r] + bia1;
  }
}

// ---------------------------------------------------------------------------
extern "C" void kernel_launch(void* const* d_in, const int* in_sizes, int n_in,
                              void* d_out, int out_size, void* d_ws, size_t ws_size,
                              hipStream_t stream) {
  (void)in_sizes; (void)n_in; (void)out_size; (void)ws_size;
  const float* x  = (const float*)d_in[0];   // (B,L,D)
  const float* W  = (const float*)d_in[1];   // (H,L,L)
  const float* Wp = (const float*)d_in[2];   // (D,D)
  const float* bp = (const float*)d_in[3];   // (D,)
  float* out = (float*)d_out;                // (B,L,D) fp32

  // scratch carving (all 16B aligned): total ~66.3 MB
  char* ws = (char*)d_ws;
  size_t off = 0;
  bhalf_t* xT  = (bhalf_t*)(ws + off); off += (size_t)Bn * Dn * Ln * 2;  // 32 MB
  bhalf_t* Wpb = (bhalf_t*)(ws + off); off += (size_t)Dn * Dn * 2;       //  2 MB
  float* row_m   = (float*)(ws + off); off += (size_t)Hn * Ln * 4;
  float* row_inv = (float*)(ws + off); off += (size_t)Hn * Ln * 4;
  bhalf_t* mixb = (bhalf_t*)(ws + off); off += (size_t)Bn * Ln * Dn * 2; // 32 MB

  k_xpose<<<dim3(Ln / 32, Dn / 32, Bn), dim3(32, 32), 0, stream>>>(x, xT);
  k_wproj<<<dim3((Dn * Dn) / 256), dim3(256), 0, stream>>>(Wp, Wpb);
  k_stats<<<dim3(Hn * Ln), dim3(256), 0, stream>>>(W, row_m, row_inv);
  k_mix<<<dim3(Ln / 64, Hn), dim3(256), 0, stream>>>(W, xT, row_m, row_inv, mixb);
  k_proj<<<dim3((Bn * Ln) / 64, Dn / 64), dim3(256), 0, stream>>>(mixb, Wpb, bp, out);
}
